// NTXentLoss_9388798509767
// MI455X (gfx1250) — compile-verified, matrix operand order checked by reading, not run on hardware
//
#include <hip/hip_runtime.h>
#include <hip/hip_fp16.h>

typedef _Float16 v16h __attribute__((ext_vector_type(16)));
typedef _Float16 v8h  __attribute__((ext_vector_type(8)));
typedef float    v8f  __attribute__((ext_vector_type(8)));
typedef float    v4f  __attribute__((ext_vector_type(4)));

#define NB      8192
#define ND      256
#define INV_T   2.0f      // 1 / 0.5
#define BASE_T  0.07f
#define JCHUNKS 8
#define JT_PER  (NB / 16 / JCHUNKS)   // 64 j-tiles per chunk

// ---------------------------------------------------------------------------
// Kernel 1: L2-normalize rows, split into f16 hi + lo (hi+lo ~ 22-bit mantissa)
// one wave per row; lane handles 8 contiguous elements
// ---------------------------------------------------------------------------
__global__ __launch_bounds__(256) void k_normalize(
    const float* __restrict__ f, _Float16* __restrict__ hi, _Float16* __restrict__ lo)
{
    const int lane = threadIdx.x & 31;
    const int wave = threadIdx.x >> 5;
    const int row  = blockIdx.x * 8 + wave;

    const float* rp = f + (size_t)row * ND + lane * 8;
    v4f v0 = *(const v4f*)rp;
    v4f v1 = *(const v4f*)(rp + 4);

    float s = v0.x*v0.x + v0.y*v0.y + v0.z*v0.z + v0.w*v0.w
            + v1.x*v1.x + v1.y*v1.y + v1.z*v1.z + v1.w*v1.w;
    #pragma unroll
    for (int m = 16; m > 0; m >>= 1) s += __shfl_xor(s, m, 32);

    const float inv = 1.0f / fmaxf(sqrtf(s), 1e-12f);

    float x[8] = { v0.x, v0.y, v0.z, v0.w, v1.x, v1.y, v1.z, v1.w };
    v8h h8, l8;
    #pragma unroll
    for (int i = 0; i < 8; ++i) {
        float xv = x[i] * inv;          // |xv| <= 1, f16-safe range
        _Float16 h = (_Float16)xv;
        float    r = xv - (float)h;     // residual ~2^-12 scale
        h8[i] = h;
        l8[i] = (_Float16)r;
    }
    *(v8h*)(hi + (size_t)row * ND + lane * 8) = h8;
    *(v8h*)(lo + (size_t)row * ND + lane * 8) = l8;
}

// ---------------------------------------------------------------------------
// fragment load: 16 f16 per lane as two 16B chunks.
// WMMA 16x16x32 f16 A/B layout: lane<16 -> row lane, K {0..7,16..23};
// lane>=16 -> row lane-16, K {8..15,24..31}  (A and B identical for Gram).
// ---------------------------------------------------------------------------
__device__ __forceinline__ v16h load_frag(const _Float16* p)
{
    v8h a = *(const v8h*)p;
    v8h b = *(const v8h*)(p + 16);
    return __builtin_shufflevector(a, b, 0,1,2,3,4,5,6,7,8,9,10,11,12,13,14,15);
}

// ---------------------------------------------------------------------------
// Kernel 2: fused Gram-tile WMMA + masked row reductions
// wave w of block bx owns i-block (bx*8+w) -> 16 rows; blockIdx.y = j-chunk
// ---------------------------------------------------------------------------
__global__ __launch_bounds__(256) void k_main(
    const _Float16* __restrict__ hi, const _Float16* __restrict__ lo,
    const int* __restrict__ labels, const int* __restrict__ spk,
    float* __restrict__ pPosSum, float* __restrict__ pExpSum,
    float* __restrict__ pPosCnt, float* __restrict__ pNCnt)
{
    const int lane    = threadIdx.x & 31;
    const int wave    = threadIdx.x >> 5;
    const int lane_lo = lane & 15;
    const int hig     = lane >> 4;          // 0: rows 0-7 of tile, 1: rows 8-15
    const int i_base  = (blockIdx.x * 8 + wave) * 16;
    const int jc      = blockIdx.y;

    // --- A fragments for this wave's 16 rows, all K (resident in VGPRs) ---
    v16h aHi[8], aLo[8];
    {
        const size_t abase = (size_t)(i_base + lane_lo) * ND + hig * 8;
        #pragma unroll
        for (int kk = 0; kk < 8; ++kk) {
            aHi[kk] = load_frag(hi + abase + kk * 32);
            aLo[kk] = load_frag(lo + abase + kk * 32);
        }
    }

    int labR[8], spkR[8];
    #pragma unroll
    for (int r = 0; r < 8; ++r) {
        int row = i_base + hig * 8 + r;
        labR[r] = labels[row];
        spkR[r] = spk[row];
    }

    float nCnt[8] = {}, posCnt[8] = {}, posSum[8] = {}, expSum[8] = {};

    const int jt0 = jc * JT_PER;
    for (int jt = jt0; jt < jt0 + JT_PER; ++jt) {
        const int jbase = jt * 16;
        const size_t bbase = (size_t)(jbase + lane_lo) * ND + hig * 8;

        v8f c = {};
        #pragma unroll
        for (int kk = 0; kk < 8; ++kk) {
            v16h bH = load_frag(hi + bbase + kk * 32);
            v16h bL = load_frag(lo + bbase + kk * 32);
            c = __builtin_amdgcn_wmma_f32_16x16x32_f16(false, aHi[kk], false, bH, (short)0, c, false, false);
            c = __builtin_amdgcn_wmma_f32_16x16x32_f16(false, aHi[kk], false, bL, (short)0, c, false, false);
            c = __builtin_amdgcn_wmma_f32_16x16x32_f16(false, aLo[kk], false, bH, (short)0, c, false, false);
        }

        const int col  = jbase + lane_lo;      // this lane's column of the tile
        const int labC = labels[col];
        const int spkC = spk[col];

        #pragma unroll
        for (int r = 0; r < 8; ++r) {
            float s    = c[r] * INV_T;                 // sim/T, bounded [-2,2]
            int   rowg = i_base + hig * 8 + r;
            bool  sc   = (labC == labR[r]);
            bool  sp   = (spkC == spkR[r]);
            bool  pos  = sc && sp && (col != rowg);
            bool  neg  = sc && !sp;                    // diag has sp==true
            nCnt[r]   += sc  ? 1.0f : 0.0f;
            posCnt[r] += pos ? 1.0f : 0.0f;
            posSum[r] += pos ? s : 0.0f;
            expSum[r] += neg ? __expf(s) : 0.0f;       // no max-shift needed
        }
    }

    // reduce across the 16 lanes that share the same rows (fixed order -> deterministic)
    #pragma unroll
    for (int r = 0; r < 8; ++r) {
        #pragma unroll
        for (int m = 1; m < 16; m <<= 1) {
            nCnt[r]   += __shfl_xor(nCnt[r],   m, 32);
            posCnt[r] += __shfl_xor(posCnt[r], m, 32);
            posSum[r] += __shfl_xor(posSum[r], m, 32);
            expSum[r] += __shfl_xor(expSum[r], m, 32);
        }
    }

    if (lane_lo == 0) {
        #pragma unroll
        for (int r = 0; r < 8; ++r) {
            int row = i_base + hig * 8 + r;
            int idx = jc * NB + row;
            pNCnt[idx]   = nCnt[r];
            pPosCnt[idx] = posCnt[r];
            pPosSum[idx] = posSum[r];
            pExpSum[idx] = expSum[r];
        }
    }
}

// ---------------------------------------------------------------------------
// Kernel 3: combine j-chunk partials, per-row loss terms, deterministic reduce
// ---------------------------------------------------------------------------
__global__ __launch_bounds__(1024) void k_finalize(
    const float* __restrict__ pPosSum, const float* __restrict__ pExpSum,
    const float* __restrict__ pPosCnt, const float* __restrict__ pNCnt,
    float* __restrict__ out)
{
    __shared__ float sNum[1024];
    __shared__ float sDen[1024];

    float num = 0.0f, den = 0.0f;
    for (int row = threadIdx.x; row < NB; row += 1024) {
        float n = 0.0f, pc = 0.0f, ps = 0.0f, es = 0.0f;
        #pragma unroll
        for (int jcu = 0; jcu < JCHUNKS; ++jcu) {
            int idx = jcu * NB + row;
            n  += pNCnt[idx];
            pc += pPosCnt[idx];
            ps += pPosSum[idx];
            es += pExpSum[idx];
        }
        float pos_sim = ps / (pc + 1e-8f);
        float neg_sim = logf(es + pc + 1.0f);   // + exp(0) terms (diag + positives)
        float valid   = (n > 1.0f) ? 1.0f : 0.0f;
        num += valid * (pos_sim - neg_sim) / n;
        den += valid / n;
    }
    sNum[threadIdx.x] = num;
    sDen[threadIdx.x] = den;
    __syncthreads();
    for (int s = 512; s > 0; s >>= 1) {
        if (threadIdx.x < s) {
            sNum[threadIdx.x] += sNum[threadIdx.x + s];
            sDen[threadIdx.x] += sDen[threadIdx.x + s];
        }
        __syncthreads();
    }
    if (threadIdx.x == 0)
        out[0] = (sDen[0] > 0.0f) ? (-sNum[0] / sDen[0] * BASE_T) : 0.0f;
}

// ---------------------------------------------------------------------------
extern "C" void kernel_launch(void* const* d_in, const int* in_sizes, int n_in,
                              void* d_out, int out_size, void* d_ws, size_t ws_size,
                              hipStream_t stream)
{
    const float* feat   = (const float*)d_in[0];
    const int*   labels = (const int*)d_in[1];
    const int*   spk    = (const int*)d_in[2];
    float*       out    = (float*)d_out;

    char* w = (char*)d_ws;
    _Float16* hi = (_Float16*)(w);                             // 4 MB
    _Float16* lo = (_Float16*)(w + (size_t)NB * ND * 2);       // 4 MB
    char* pbase   = w + (size_t)NB * ND * 4;
    float* pPosSum = (float*)(pbase);
    float* pExpSum = (float*)(pbase + 1 * JCHUNKS * NB * sizeof(float));
    float* pPosCnt = (float*)(pbase + 2 * JCHUNKS * NB * sizeof(float));
    float* pNCnt   = (float*)(pbase + 3 * JCHUNKS * NB * sizeof(float));

    k_normalize<<<NB / 8, 256, 0, stream>>>(feat, hi, lo);

    dim3 grid(NB / 16 / 8, JCHUNKS, 1);   // 64 x 8 blocks, 8 waves each
    k_main<<<grid, 256, 0, stream>>>(hi, lo, labels, spk,
                                     pPosSum, pExpSum, pPosCnt, pNCnt);

    k_finalize<<<1, 1024, 0, stream>>>(pPosSum, pExpSum, pPosCnt, pNCnt, out);
}
